// DecoderV2_14199161880700
// MI455X (gfx1250) — matrix operand order repaired and checked by reading, not compiled, not run on hardware
//
#include <hip/hip_runtime.h>
#include <hip/hip_bf16.h>

// ---------------------------------------------------------------------------
// CDNA5 (gfx1250) decoder: all GEMMs via v_wmma_f32_16x16x32_bf16 (wave32).
// Sequential T=32 recurrence -> per-step kernel chain on `stream`.
// Sizes (from reference): B=64 T=32 S=400 E=300 EH=DH=A=512 V=50000 P=2
// ---------------------------------------------------------------------------

#define B_  64
#define T_  32
#define S_  400
#define E_  300
#define EH_ 512
#define DH_ 512
#define A_  512
#define V_  50000
#define KCAT 1324           // E + EH + DH (= E + DH + EH)
#define KPAD 1344           // KCAT padded to multiple of 32 (and of 64)

typedef __bf16 bf16_t;
typedef __attribute__((ext_vector_type(16))) __bf16 v16bf;
typedef __attribute__((ext_vector_type(8)))  float  v8f;

union BFrag { v16bf v; unsigned int u[8]; };

static __device__ __forceinline__ unsigned int pack2bf(float a, float b) {
    union { __bf16 h; unsigned short u; } ua, ub;
    ua.h = (__bf16)a; ub.h = (__bf16)b;
    return (unsigned int)ua.u | ((unsigned int)ub.u << 16);
}

static __device__ __forceinline__ float sigmoidf_(float x) {
    return 1.0f / (1.0f + __expf(-x));
}

// Load a 16-value bf16 B fragment (two 16B chunks) from a weight row.
static __device__ __forceinline__ BFrag load_bfrag(const bf16_t* __restrict__ p) {
    const uint4 q0 = *(const uint4*)(p);
    const uint4 q1 = *(const uint4*)(p + 8);
    BFrag f;
    f.u[0] = q0.x; f.u[1] = q0.y; f.u[2] = q0.z; f.u[3] = q0.w;
    f.u[4] = q1.x; f.u[5] = q1.y; f.u[6] = q1.z; f.u[7] = q1.w;
    return f;
}

// ---------------------------------------------------------------------------
// Generic WMMA GEMM:  C[M,N] = A_f32[M,K] @ Bw_bf16[N,K]^T (+ bias[N])
//   - A is f32 in memory, converted to bf16 in-register
//   - Bw is bf16 weights, row n contiguous along K (ldb = K, even)
//   - block = 128 threads = 4 waves; wave tile = 16(M) x 64(N); block = 16x256
//   - requires: M % 16 == 0, K % 64 == 0, N % 16 == 0 (true for all call sites)
//   - N-boundary subtiles: B row index is CLAMPED so loads stay in-bounds and
//     the k-loop is branch-free (EXEC stays all-1s around every WMMA); only
//     the epilogue stores are guarded.
//   - All state in named scalars (no arrays) so pointers stay SSA: the
//     backend keeps global_load_b128 (LOADcnt only, no flat/DScnt coupling)
//     and can clause all 8 B loads ahead of the 4 WMMAs.
// ---------------------------------------------------------------------------
__global__ __launch_bounds__(128)
void gemm_wmma_bf16(float* __restrict__ C, int ldc,
                    const float* __restrict__ A, int lda,
                    const bf16_t* __restrict__ Bw, int ldb,
                    const float* __restrict__ bias,
                    int M, int N, int K)
{
    const int wave   = threadIdx.x >> 5;
    const int lane   = threadIdx.x & 31;
    const int lhalf  = lane >> 4;         // 0 = lanes 0-15, 1 = lanes 16-31
    const int l16    = lane & 15;
    const int mtile  = blockIdx.y * 16;
    const int ntile0 = blockIdx.x * 256 + wave * 64;

    v8f acc0 = {}, acc1 = {}, acc2 = {}, acc3 = {};

    // A row pointer (per lane): lo lanes K=+0..15 contiguous, hi lanes +16..+31
    const float* pA = A + (size_t)(mtile + l16) * lda + lhalf * 16;

    // B row pointers per 16-col subtile; clamp invalid subtiles to row l16
    // so loads are always in-bounds (results discarded at store time).
    const int  n00 = ntile0,      n01 = ntile0 + 16;
    const int  n02 = ntile0 + 32, n03 = ntile0 + 48;
    const bool nv0 = n00 < N, nv1 = n01 < N, nv2 = n02 < N, nv3 = n03 < N;
    // B 32x16 fragment: lo lanes K=k0..k0+15 contiguous, hi lanes +16..+31
    const bf16_t* pB0 = Bw + (size_t)(nv0 ? n00 + l16 : l16) * ldb + lhalf * 16;
    const bf16_t* pB1 = Bw + (size_t)(nv1 ? n01 + l16 : l16) * ldb + lhalf * 16;
    const bf16_t* pB2 = Bw + (size_t)(nv2 ? n02 + l16 : l16) * ldb + lhalf * 16;
    const bf16_t* pB3 = Bw + (size_t)(nv3 ? n03 + l16 : l16) * ldb + lhalf * 16;

    #pragma unroll 2
    for (int k0 = 0; k0 < K; k0 += 32) {
        // ---- A fragment ----
        BFrag af;
        {
            const float4 f0 = *(const float4*)(pA);
            const float4 f1 = *(const float4*)(pA + 4);
            const float4 f2 = *(const float4*)(pA + 8);
            const float4 f3 = *(const float4*)(pA + 12);
            af.u[0] = pack2bf(f0.x, f0.y); af.u[1] = pack2bf(f0.z, f0.w);
            af.u[2] = pack2bf(f1.x, f1.y); af.u[3] = pack2bf(f1.z, f1.w);
            af.u[4] = pack2bf(f2.x, f2.y); af.u[5] = pack2bf(f2.z, f2.w);
            af.u[6] = pack2bf(f3.x, f3.y); af.u[7] = pack2bf(f3.z, f3.w);
            pA += 32;
        }
        // ---- all 4 B fragments first (one big load clause) ----
        const BFrag b0 = load_bfrag(pB0); pB0 += 32;
        const BFrag b1 = load_bfrag(pB1); pB1 += 32;
        const BFrag b2 = load_bfrag(pB2); pB2 += 32;
        const BFrag b3 = load_bfrag(pB3); pB3 += 32;
        // ---- 4 WMMAs ----
        acc0 = __builtin_amdgcn_wmma_f32_16x16x32_bf16(
            false, af.v, false, b0.v, (short)0, acc0, false, false);
        acc1 = __builtin_amdgcn_wmma_f32_16x16x32_bf16(
            false, af.v, false, b1.v, (short)0, acc1, false, false);
        acc2 = __builtin_amdgcn_wmma_f32_16x16x32_bf16(
            false, af.v, false, b2.v, (short)0, acc2, false, false);
        acc3 = __builtin_amdgcn_wmma_f32_16x16x32_bf16(
            false, af.v, false, b3.v, (short)0, acc3, false, false);
    }

    // ---- epilogue: VGPR r -> M = mtile + r + 8*lhalf, col = n0 + l16 ----
    if (nv0) {
        const int n = n00 + l16;
        const float bv = bias ? bias[n] : 0.0f;
        #pragma unroll
        for (int r = 0; r < 8; ++r)
            C[(size_t)(mtile + r + 8 * lhalf) * ldc + n] = acc0[r] + bv;
    }
    if (nv1) {
        const int n = n01 + l16;
        const float bv = bias ? bias[n] : 0.0f;
        #pragma unroll
        for (int r = 0; r < 8; ++r)
            C[(size_t)(mtile + r + 8 * lhalf) * ldc + n] = acc1[r] + bv;
    }
    if (nv2) {
        const int n = n02 + l16;
        const float bv = bias ? bias[n] : 0.0f;
        #pragma unroll
        for (int r = 0; r < 8; ++r)
            C[(size_t)(mtile + r + 8 * lhalf) * ldc + n] = acc2[r] + bv;
    }
    if (nv3) {
        const int n = n03 + l16;
        const float bv = bias ? bias[n] : 0.0f;
        #pragma unroll
        for (int r = 0; r < 8; ++r)
            C[(size_t)(mtile + r + 8 * lhalf) * ldc + n] = acc3[r] + bv;
    }
}

// ---------------------------------------------------------------------------
// Setup / pointwise kernels
// ---------------------------------------------------------------------------
__global__ void zero_f32(float* p, int n) {
    int i = blockIdx.x * blockDim.x + threadIdx.x;
    if (i < n) p[i] = 0.0f;
}

__global__ void add_vec(float* dst, const float* a, const float* b, int n) {
    int i = blockIdx.x * blockDim.x + threadIdx.x;
    if (i < n) dst[i] = a[i] + b[i];
}

// f32 [N,K] -> bf16 [N,Kpad] (zero pad)
__global__ void cvt_pad_bf16(bf16_t* dst, const float* src, int N, int K, int Kpad) {
    long long i = (long long)blockIdx.x * blockDim.x + threadIdx.x;
    long long total = (long long)N * Kpad;
    if (i >= total) return;
    int n = (int)(i / Kpad), k = (int)(i % Kpad);
    dst[i] = (k < K) ? (bf16_t)src[(size_t)n * K + k] : (bf16_t)0.0f;
}

// Wg [2048, KPAD] = [ W_ih (K=812) | W_hh (K=512) | 0-pad ]
__global__ void build_wg(bf16_t* dst, const float* wih, const float* whh) {
    long long i = (long long)blockIdx.x * blockDim.x + threadIdx.x;
    long long total = (long long)(4 * DH_) * KPAD;
    if (i >= total) return;
    int n = (int)(i / KPAD), k = (int)(i % KPAD);
    float v = 0.0f;
    if (k < E_ + EH_)      v = wih[(size_t)n * (E_ + EH_) + k];
    else if (k < KCAT)     v = whh[(size_t)n * DH_ + (k - E_ - EH_)];
    dst[i] = (bf16_t)v;
}

// dst[b, 0:KPAD] = [s0(l0) | s1(l1) | s2(l2) | 0-pad], per-source row strides
__global__ void concat3_pad(float* dst,
                            const float* s0, int l0, int str0,
                            const float* s1, int l1, int str1,
                            const float* s2, int l2, int str2) {
    int b = blockIdx.x;
    for (int k = threadIdx.x; k < KPAD; k += blockDim.x) {
        float v = 0.0f;
        if (k < l0)                v = s0[(size_t)b * str0 + k];
        else if (k < l0 + l1)      v = s1[(size_t)b * str1 + (k - l0)];
        else if (k < l0 + l1 + l2) v = s2[(size_t)b * str2 + (k - l0 - l1)];
        dst[(size_t)b * KPAD + k] = v;
    }
}

// gates[B,4*DH] (i,f,g,o) -> update h,c in place
__global__ void lstm_pointwise(const float* __restrict__ gates,
                               float* __restrict__ h, float* __restrict__ c) {
    int idx = blockIdx.x * blockDim.x + threadIdx.x;
    if (idx >= B_ * DH_) return;
    int b = idx >> 9, d = idx & (DH_ - 1);
    const float* g = gates + (size_t)b * (4 * DH_);
    float ig = sigmoidf_(g[d]);
    float fg = sigmoidf_(g[DH_ + d]);
    float gg = tanhf(g[2 * DH_ + d]);
    float og = sigmoidf_(g[3 * DH_ + d]);
    float cn = fg * c[idx] + ig * gg;
    c[idx] = cn;
    h[idx] = og * tanhf(cn);
}

// Fused attention: energy(+mask) -> softmax -> context, coverage update,
// and emits attn/coverage(pre-update)/energy for this timestep.
// One block (256 thr) per batch row b.
__global__ __launch_bounds__(256)
void attend_kernel(const float* __restrict__ pre,   // [B,S,A]
                   const float* __restrict__ q,     // [B,A]
                   const float* __restrict__ w_v,   // [A]
                   const float* __restrict__ w_cov, // [A]
                   const int*   __restrict__ mask,  // [B,S]
                   const float* __restrict__ enc,   // [B,S,EH]
                   float* __restrict__ cov,         // [B,S] in/out
                   float* __restrict__ ctx,         // [B,EH] out
                   float* __restrict__ attn_out,    // [B,S] (at +t*B*S)
                   float* __restrict__ cov_out,     // [B,S]
                   float* __restrict__ en_out)      // [B,S]
{
    __shared__ float sq[A_], swv[A_], swc[A_];
    __shared__ float sattn[S_];
    __shared__ float redmax[8], redsum[8];

    const int b = blockIdx.x, tid = threadIdx.x;
    for (int a = tid; a < A_; a += 256) {
        sq[a]  = q[(size_t)b * A_ + a];
        swv[a] = w_v[a];
        swc[a] = w_cov[a];
    }
    __syncthreads();

    // energies + mask
    for (int s = tid; s < S_; s += 256) {
        const float cv = cov[(size_t)b * S_ + s];
        const float* pr = pre + ((size_t)b * S_ + s) * A_;
        float e = 0.0f;
        #pragma unroll 4
        for (int a = 0; a < A_; ++a)
            e += swv[a] * tanhf(pr[a] + sq[a] + cv * swc[a]);
        if (mask[(size_t)b * S_ + s] == 1) e = -1.0e12f;
        en_out[(size_t)b * S_ + s]  = e;
        cov_out[(size_t)b * S_ + s] = cv;   // coverage BEFORE update (ref semantics)
        sattn[s] = e;
    }
    __syncthreads();

    // softmax: block max
    float lm = -3.0e38f;
    for (int s = tid; s < S_; s += 256) lm = fmaxf(lm, sattn[s]);
    for (int off = 16; off; off >>= 1) lm = fmaxf(lm, __shfl_xor(lm, off));
    if ((tid & 31) == 0) redmax[tid >> 5] = lm;
    __syncthreads();
    if (tid == 0) {
        float m = redmax[0];
        for (int i = 1; i < 8; ++i) m = fmaxf(m, redmax[i]);
        redmax[0] = m;
    }
    __syncthreads();
    const float gmax = redmax[0];

    float ls = 0.0f;
    for (int s = tid; s < S_; s += 256) {
        float e = __expf(sattn[s] - gmax);
        sattn[s] = e;
        ls += e;
    }
    for (int off = 16; off; off >>= 1) ls += __shfl_xor(ls, off);
    if ((tid & 31) == 0) redsum[tid >> 5] = ls;
    __syncthreads();
    if (tid == 0) {
        float ssum = 0.0f;
        for (int i = 0; i < 8; ++i) ssum += redsum[i];
        redsum[0] = ssum;
    }
    __syncthreads();
    const float inv = 1.0f / redsum[0];

    for (int s = tid; s < S_; s += 256) {
        float a = sattn[s] * inv;
        sattn[s] = a;
        attn_out[(size_t)b * S_ + s] = a;
        cov[(size_t)b * S_ + s] += a;
    }
    __syncthreads();

    // context[b,d] = sum_s attn[s] * enc[b,s,d]  (coalesced over d)
    for (int d = tid; d < EH_; d += 256) {
        const float* ep = enc + (size_t)b * S_ * EH_ + d;
        float acc = 0.0f;
        for (int s = 0; s < S_; ++s) acc += sattn[s] * ep[(size_t)s * EH_];
        ctx[(size_t)b * EH_ + d] = acc;
    }
}

// maxout over P=2 pairs: maxo[b,j] = max(readout[b,2j], readout[b,2j+1])
__global__ void maxout_kernel(const float* __restrict__ readout,
                              float* __restrict__ maxo) {
    int idx = blockIdx.x * blockDim.x + threadIdx.x;
    if (idx >= B_ * (DH_ / 2)) return;
    int b = idx >> 8, j = idx & 255;
    float a  = readout[(size_t)b * DH_ + 2 * j];
    float bb = readout[(size_t)b * DH_ + 2 * j + 1];
    maxo[idx] = fmaxf(a, bb);
}

// ---------------------------------------------------------------------------
// Host-side launcher
// ---------------------------------------------------------------------------
extern "C" void kernel_launch(void* const* d_in, const int* in_sizes, int n_in,
                              void* d_out, int out_size, void* d_ws, size_t ws_size,
                              hipStream_t stream) {
    const float* trg     = (const float*)d_in[0];   // [B,T,E]
    const float* enc_h   = (const float*)d_in[2];   // [1,B,EH]
    const float* enc_c   = (const float*)d_in[3];
    const float* enc_out = (const float*)d_in[4];   // [B,S,EH]
    const int*   emask   = (const int*)  d_in[5];
    const float* W_trans = (const float*)d_in[6];
    const float* b_trans = (const float*)d_in[7];
    const float* W_ih    = (const float*)d_in[8];
    const float* W_hh    = (const float*)d_in[9];
    const float* b_ih    = (const float*)d_in[10];
    const float* b_hh    = (const float*)d_in[11];
    const float* W_pre   = (const float*)d_in[12];
    const float* b_pre   = (const float*)d_in[13];
    const float* W_q     = (const float*)d_in[14];
    const float* w_v     = (const float*)d_in[15];
    const float* w_cov   = (const float*)d_in[16];
    const float* W_read  = (const float*)d_in[17];
    const float* b_read  = (const float*)d_in[18];
    const float* W_logit = (const float*)d_in[19];
    const float* b_logit = (const float*)d_in[20];

    float* out = (float*)d_out;
    float* out_logits = out;                                   // [T,B,V]
    float* out_attn   = out + (size_t)T_ * B_ * V_;            // [T,B,S]
    float* out_cov    = out_attn + (size_t)T_ * B_ * S_;
    float* out_en     = out_cov  + (size_t)T_ * B_ * S_;

    // ---- workspace layout (256B aligned) ----
    char* base = (char*)d_ws;
    size_t off = 0;
    auto alloc = [&](size_t bytes) {
        size_t o = off;
        off = (off + bytes + 255) & ~(size_t)255;
        return o;
    };
    float*  pre      = (float*) (base + alloc((size_t)B_ * S_ * A_ * 4));   // 52.4 MB
    bf16_t* WgBf     = (bf16_t*)(base + alloc((size_t)4 * DH_ * KPAD * 2)); // 5.5 MB
    bf16_t* WreadBf  = (bf16_t*)(base + alloc((size_t)DH_ * KPAD * 2));
    bf16_t* WqBf     = (bf16_t*)(base + alloc((size_t)A_ * DH_ * 2));
    bf16_t* WlogitBf = (bf16_t*)(base + alloc((size_t)V_ * (DH_/2) * 2));   // 25.6 MB
    bf16_t* WpreBf   = (bf16_t*)(base + alloc((size_t)A_ * EH_ * 2));
    bf16_t* WtransBf = (bf16_t*)(base + alloc((size_t)DH_ * EH_ * 2));
    float*  gbias    = (float*) (base + alloc((size_t)4 * DH_ * 4));
    float*  h        = (float*) (base + alloc((size_t)B_ * DH_ * 4));
    float*  c        = (float*) (base + alloc((size_t)B_ * DH_ * 4));
    float*  ctx      = (float*) (base + alloc((size_t)B_ * EH_ * 4));
    float*  cov      = (float*) (base + alloc((size_t)B_ * S_ * 4));
    float*  xh1      = (float*) (base + alloc((size_t)B_ * KPAD * 4));
    float*  xh2      = (float*) (base + alloc((size_t)B_ * KPAD * 4));
    float*  gates    = (float*) (base + alloc((size_t)B_ * 4 * DH_ * 4));
    float*  qbuf     = (float*) (base + alloc((size_t)B_ * A_ * 4));
    float*  rdout    = (float*) (base + alloc((size_t)B_ * DH_ * 4));
    float*  maxo     = (float*) (base + alloc((size_t)B_ * (DH_/2) * 4));
    (void)ws_size; (void)in_sizes; (void)n_in; (void)out_size;

    auto cdiv = [](long long a, long long b) { return (int)((a + b - 1) / b); };

    // ---- phase 0: state init + weight conversion ----
    zero_f32<<<cdiv(B_ * EH_, 256), 256, 0, stream>>>(ctx, B_ * EH_);
    zero_f32<<<cdiv(B_ * S_, 256), 256, 0, stream>>>(cov, B_ * S_);
    add_vec<<<cdiv(4 * DH_, 256), 256, 0, stream>>>(gbias, b_ih, b_hh, 4 * DH_);

    build_wg<<<cdiv((long long)4 * DH_ * KPAD, 256), 256, 0, stream>>>(WgBf, W_ih, W_hh);
    cvt_pad_bf16<<<cdiv((long long)DH_ * KPAD, 256), 256, 0, stream>>>(WreadBf, W_read, DH_, KCAT, KPAD);
    cvt_pad_bf16<<<cdiv((long long)A_ * DH_, 256), 256, 0, stream>>>(WqBf, W_q, A_, DH_, DH_);
    cvt_pad_bf16<<<cdiv((long long)V_ * (DH_/2), 256), 256, 0, stream>>>(WlogitBf, W_logit, V_, DH_/2, DH_/2);
    cvt_pad_bf16<<<cdiv((long long)A_ * EH_, 256), 256, 0, stream>>>(WpreBf, W_pre, A_, EH_, EH_);
    cvt_pad_bf16<<<cdiv((long long)DH_ * EH_, 256), 256, 0, stream>>>(WtransBf, W_trans, DH_, EH_, EH_);

    const dim3 gemmBlk(128);
    // h0 = enc_h[0] @ W_trans^T + b_trans ; c0 likewise
    gemm_wmma_bf16<<<dim3(cdiv(DH_, 256), B_ / 16), gemmBlk, 0, stream>>>(
        h, DH_, enc_h, EH_, WtransBf, EH_, b_trans, B_, DH_, EH_);
    gemm_wmma_bf16<<<dim3(cdiv(DH_, 256), B_ / 16), gemmBlk, 0, stream>>>(
        c, DH_, enc_c, EH_, WtransBf, EH_, b_trans, B_, DH_, EH_);
    // pre_compute = enc_out @ W_pre^T + b_pre : [B*S, A]
    gemm_wmma_bf16<<<dim3(cdiv(A_, 256), (B_ * S_) / 16), gemmBlk, 0, stream>>>(
        pre, A_, enc_out, EH_, WpreBf, EH_, b_pre, B_ * S_, A_, EH_);

    // ---- recurrent steps ----
    for (int t = 0; t < T_; ++t) {
        const float* emb_t = trg + (size_t)t * E_;   // row stride T*E

        // x = [emb, pre_context, h] -> gates GEMM input
        concat3_pad<<<B_, 256, 0, stream>>>(xh1,
            emb_t, E_, T_ * E_, ctx, EH_, EH_, h, DH_, DH_);
        gemm_wmma_bf16<<<dim3(cdiv(4 * DH_, 256), B_ / 16), gemmBlk, 0, stream>>>(
            gates, 4 * DH_, xh1, KPAD, WgBf, KPAD, gbias, B_, 4 * DH_, KPAD);
        lstm_pointwise<<<cdiv(B_ * DH_, 256), 256, 0, stream>>>(gates, h, c);

        // q = h_new @ W_q^T
        gemm_wmma_bf16<<<dim3(cdiv(A_, 256), B_ / 16), gemmBlk, 0, stream>>>(
            qbuf, A_, h, DH_, WqBf, DH_, nullptr, B_, A_, DH_);

        attend_kernel<<<B_, 256, 0, stream>>>(
            pre, qbuf, w_v, w_cov, emask, enc_out, cov, ctx,
            out_attn + (size_t)t * B_ * S_,
            out_cov  + (size_t)t * B_ * S_,
            out_en   + (size_t)t * B_ * S_);

        // readout = [emb, h_new, context] @ W_read^T + b_read
        concat3_pad<<<B_, 256, 0, stream>>>(xh2,
            emb_t, E_, T_ * E_, h, DH_, DH_, ctx, EH_, EH_);
        gemm_wmma_bf16<<<dim3(cdiv(DH_, 256), B_ / 16), gemmBlk, 0, stream>>>(
            rdout, DH_, xh2, KPAD, WreadBf, KPAD, b_read, B_, DH_, KPAD);
        maxout_kernel<<<cdiv(B_ * (DH_/2), 256), 256, 0, stream>>>(rdout, maxo);

        // logits_t = maxo @ W_logit^T + b_logit  -> directly into d_out
        gemm_wmma_bf16<<<dim3(cdiv(V_, 256), B_ / 16), gemmBlk, 0, stream>>>(
            out_logits + (size_t)t * B_ * V_, V_, maxo, DH_/2, WlogitBf, DH_/2,
            b_logit, B_, V_, DH_/2);
    }
}
